// DeepSet_72894184947809
// MI455X (gfx1250) — matrix operand order repaired and checked by reading, not compiled, not run on hardware
//
#include <hip/hip_runtime.h>

// ---------------------------------------------------------------------------
// DeepSet fused pipeline for MI455X (gfx1250, wave32, WMMA 16x16x32 f16).
// Compute-bound (~84 GFLOP vs ~120MB traffic) -> all matmuls on v_wmma.
// Each wave owns one fixed N-tile (nt == wid): perfect 8-wave balance and
// weight (B) fragments loaded once per layer per wave.
// ---------------------------------------------------------------------------

#define BATCH   16384
#define HDIM    128
#define TB      16        // batch rows per workgroup in k_mlp1
#define RB2     2         // batch rows per workgroup in k_mlp2

typedef __attribute__((ext_vector_type(16))) _Float16 v16h;
typedef __attribute__((ext_vector_type(8)))  _Float16 v8h;
typedef __attribute__((ext_vector_type(8)))  float    v8f;
typedef int v4i_vs __attribute__((vector_size(16)));   // matches builtin param

#if defined(__HIP_DEVICE_COMPILE__) && __has_builtin(__builtin_amdgcn_global_load_async_to_lds_b128)
#define HAVE_ASYNC_G2L 1
#endif

__device__ __forceinline__ v8f wmma16(v16h a, v16h b, v8f c) {
  return __builtin_amdgcn_wmma_f32_16x16x32_f16(
      /*neg_a=*/false, a, /*neg_b=*/false, b,
      /*c_mod=*/(short)0, c, /*reuse_a=*/false, /*reuse_b=*/false);
}

// A-matrix 16x32 f16 fragment (ISA 7.12.2): lanes 0-15 hold K{0..7,16..23},
// lanes 16-31 hold K{8..15,24..31}, M = lane&15.
__device__ __forceinline__ v16h fragA(const _Float16* base, int row0, int stride,
                                      int k0, int lane) {
  int r  = row0 + (lane & 15);
  int kb = k0 + ((lane >> 4) << 3);
  const _Float16* p = base + r * stride + kb;
  v8h lo = *(const v8h*)(p);
  v8h hi = *(const v8h*)(p + 16);
  v16h o;
#pragma unroll
  for (int i = 0; i < 8; ++i) { o[i] = lo[i]; o[i + 8] = hi[i]; }
  return o;
}

// A fragment for K=16 (layer 1), zero-padded to K=32 (branchless).
__device__ __forceinline__ v16h fragA_K16(const _Float16* base, int row0, int lane) {
  int r  = row0 + (lane & 15);
  int kb = (lane >> 4) << 3;
  v8h lo = *(const v8h*)(base + r * 16 + kb);
  v16h o;
#pragma unroll
  for (int i = 0; i < 8; ++i) { o[i] = lo[i]; o[i + 8] = (_Float16)0.0f; }
  return o;
}

// B-matrix 32x16 f16 fragment from N-major (transposed) weights:
// lane&15 = N column; lanes 0-15 hold K{0..7,16..23}, lanes 16-31 hold K{8..15,24..31}.
__device__ __forceinline__ v16h fragB(const _Float16* base, int n0, int stride,
                                      int k0, int lane) {
  int n  = n0 + (lane & 15);
  int kb = k0 + ((lane >> 4) << 3);
  const _Float16* p = base + n * stride + kb;
  v8h lo = *(const v8h*)(p);
  v8h hi = *(const v8h*)(p + 16);
  v16h o;
#pragma unroll
  for (int i = 0; i < 8; ++i) { o[i] = lo[i]; o[i + 8] = hi[i]; }
  return o;
}

// B fragment for K=16: lanes 0-15 low half K0..7, lanes 16-31 low half K8..15;
// high half meets A's zero padding, so it may hold anything.
__device__ __forceinline__ v16h fragB_K16(const _Float16* base, int n0, int lane) {
  int n = n0 + (lane & 15);
  v8h lo = *(const v8h*)(base + n * 16 + ((lane >> 4) << 3));
  v8h hi = *(const v8h*)(base + n * 16 + 8);
  v16h o;
#pragma unroll
  for (int i = 0; i < 8; ++i) { o[i] = lo[i]; o[i + 8] = hi[i]; }
  return o;
}

// C/D layout: lane&15 = N, VGPR i -> M = i + (lane>=16 ? 8 : 0).
__device__ __forceinline__ void store_bnrelu_h(v8f acc, _Float16* dst, int row0,
                                               int n0, const float* sS,
                                               const float* sT, int lane) {
  int n  = n0 + (lane & 15);
  int mb = (lane >> 4) << 3;
  float sc = sS[n], bi = sT[n];
#pragma unroll
  for (int i = 0; i < 8; ++i) {
    float v = fmaxf(acc[i] * sc + bi, 0.0f);
    dst[(row0 + mb + i) * HDIM + n] = (_Float16)v;
  }
}

// Generic 16B copy (store path or async fallback).
__device__ __forceinline__ void copy_half8(_Float16* dst, const _Float16* src,
                                           int n_halfs, int tid, int nthreads) {
  int chunks = n_halfs >> 3;
  for (int i = tid; i < chunks; i += nthreads)
    *(v8h*)(dst + (i << 3)) = *(const v8h*)(src + (i << 3));
}

// Global -> LDS staging; async path on gfx1250 when the builtin exists.
__device__ __forceinline__ void copy_g2l(_Float16* dst, const _Float16* src,
                                         int n_halfs, int tid, int nthreads) {
#ifdef HAVE_ASYNC_G2L
  typedef __attribute__((address_space(1))) v4i_vs* gvp_t;
  typedef __attribute__((address_space(3))) v4i_vs* lvp_t;
  int chunks = n_halfs >> 3;
  for (int i = tid; i < chunks; i += nthreads)
    __builtin_amdgcn_global_load_async_to_lds_b128(
        (gvp_t)(unsigned long long)(src + (i << 3)),
        (lvp_t)(unsigned int)(unsigned long long)(dst + (i << 3)), 0, 0);
#else
  copy_half8(dst, src, n_halfs, tid, nthreads);
#endif
}

__device__ __forceinline__ void g2l_commit() {
#ifdef HAVE_ASYNC_G2L
  asm volatile("s_wait_asynccnt 0" ::: "memory");
#endif
}

// ---------------------------------------------------------------------------
// Prep: transpose weights to N-major f16, fold BN into (scale, bias).
// wT layout (halfs): w1T[2048] | w1r0T[16384] | w1r1T[16384] |
//                    w2fT[16384] | w2r0T[16384] | w2r1T[16384]   (83968 total)
// st layout (floats): bn1 {s0,t0,s1,t1,s2,t2}*128 | bn2 same      (1536 total)
// ---------------------------------------------------------------------------
__global__ __launch_bounds__(256) void k_prep(
    const float* __restrict__ w1_first, const float* __restrict__ w1_rest,
    const float* __restrict__ bn1, const float* __restrict__ w2_first,
    const float* __restrict__ w2_rest, const float* __restrict__ bn2,
    _Float16* __restrict__ wT, float* __restrict__ st)
{
  int idx = blockIdx.x * 256 + threadIdx.x;
  if (idx < 2048) {
    int n = idx >> 4, k = idx & 15;
    wT[idx] = (_Float16)w1_first[k * 128 + n];
  } else if (idx < 18432) {
    int i = idx - 2048; int n = i >> 7, k = i & 127;
    wT[idx] = (_Float16)w1_rest[k * 128 + n];
  } else if (idx < 34816) {
    int i = idx - 18432; int n = i >> 7, k = i & 127;
    wT[idx] = (_Float16)w1_rest[16384 + k * 128 + n];
  } else if (idx < 51200) {
    int i = idx - 34816; int n = i >> 7, k = i & 127;
    wT[idx] = (_Float16)w2_first[k * 128 + n];
  } else if (idx < 67584) {
    int i = idx - 51200; int n = i >> 7, k = i & 127;
    wT[idx] = (_Float16)w2_rest[k * 128 + n];
  } else if (idx < 83968) {
    int i = idx - 67584; int n = i >> 7, k = i & 127;
    wT[idx] = (_Float16)w2_rest[16384 + k * 128 + n];
  } else if (idx < 85504) {
    int j = idx - 83968;                 // 0..1535
    int c = j & 127;
    int grp = j >> 7;                    // 0..11
    const float* bn = (grp < 6) ? bn1 : bn2;
    int g2 = grp % 6;
    int layer = g2 >> 1;
    int is_t  = g2 & 1;
    float ga = bn[(layer * 4 + 0) * 128 + c];
    float be = bn[(layer * 4 + 1) * 128 + c];
    float mm = bn[(layer * 4 + 2) * 128 + c];
    float vv = bn[(layer * 4 + 3) * 128 + c];
    float s = ga * rsqrtf(vv + 1.0e-3f);
    st[j] = is_t ? (be - mm * s) : s;
  }
}

// ---------------------------------------------------------------------------
// MLP1 + agg_x: one WG = 16 batch rows -> M = 160 (10 M-tiles), 8 waves.
// Wave w computes N-tile w for all 10 M-tiles (balanced, B loaded once).
// ---------------------------------------------------------------------------
__global__ __launch_bounds__(256) void k_mlp1(
    const float* __restrict__ jets, const float* __restrict__ meanj,
    const float* __restrict__ stdj, const _Float16* __restrict__ wT,
    const float* __restrict__ st1, int* __restrict__ jn_ws,
    _Float16* __restrict__ x_ws, float* __restrict__ aggx)
{
  extern __shared__ char smem[];
  _Float16* sA0 = (_Float16*)smem;          // 160*16
  _Float16* sW  = sA0 + 160 * 16;           // 2048 + 16384 + 16384
  _Float16* sHa = sW + 34816;               // 160*128
  _Float16* sHb = sHa + 160 * HDIM;         // 160*128
  float*    sST = (float*)(sHb + 160 * HDIM);  // 768
  int*      sMask = (int*)(sST + 768);      // 160
  int*      sJn   = sMask + 160;            // 16

  int tid = threadIdx.x, lane = tid & 31, wid = tid >> 5;
  int b0 = blockIdx.x * TB;
  int n0 = wid * 16;      // this wave's fixed N-tile

  copy_g2l(sW, wT, 34816, tid, 256);
  for (int i = tid; i < 768; i += 256) sST[i] = st1[i];

  if (tid < 160) {
    int r = tid / 10, j = tid - r * 10;
    const float* src = jets + ((size_t)(b0 + r) * 10 + j) * 16;
    int m = 0;
    v8h a0, a1;
#pragma unroll
    for (int k = 0; k < 8; ++k) {
      float v0 = src[k], v1 = src[k + 8];
      m |= (v0 != 0.0f) ? 1 : 0;
      m |= (v1 != 0.0f) ? 1 : 0;
      a0[k] = (_Float16)((v0 - meanj[k]) / stdj[k]);
      a1[k] = (_Float16)((v1 - meanj[k + 8]) / stdj[k + 8]);
    }
    *(v8h*)(sA0 + (r * 10 + j) * 16)     = a0;
    *(v8h*)(sA0 + (r * 10 + j) * 16 + 8) = a1;
    sMask[tid] = m;
  }
  g2l_commit();
  __syncthreads();
  if (tid < TB) {
    int c = 0;
    for (int j = 0; j < 10; ++j) c += sMask[tid * 10 + j];
    sJn[tid] = c;
    jn_ws[b0 + tid] = c;
  }
  __syncthreads();

  // Layer 1: (160x16) @ (16x128); one B fragment per wave, 10 M-tiles.
  {
    v16h bf = fragB_K16(sW, n0, lane);
#pragma unroll
    for (int t = 0; t < 10; ++t) {
      v16h a = fragA_K16(sA0, t * 16, lane);
      v8f acc = {};
      acc = wmma16(a, bf, acc);
      store_bnrelu_h(acc, sHa, t * 16, n0, sST, sST + HDIM, lane);
    }
  }
  __syncthreads();

  // Layer 2: (160x128) @ (128x128); 4 B fragments per wave, 10 M-tiles.
  {
    const _Float16* w2 = sW + 2048;
    v16h bf0 = fragB(w2, n0, HDIM, 0,  lane);
    v16h bf1 = fragB(w2, n0, HDIM, 32, lane);
    v16h bf2 = fragB(w2, n0, HDIM, 64, lane);
    v16h bf3 = fragB(w2, n0, HDIM, 96, lane);
#pragma unroll
    for (int t = 0; t < 10; ++t) {
      v8f acc = {};
      acc = wmma16(fragA(sHa, t * 16, HDIM, 0,  lane), bf0, acc);
      acc = wmma16(fragA(sHa, t * 16, HDIM, 32, lane), bf1, acc);
      acc = wmma16(fragA(sHa, t * 16, HDIM, 64, lane), bf2, acc);
      acc = wmma16(fragA(sHa, t * 16, HDIM, 96, lane), bf3, acc);
      store_bnrelu_h(acc, sHb, t * 16, n0, sST + 2 * HDIM, sST + 3 * HDIM, lane);
    }
  }
  __syncthreads();

  // Layer 3
  {
    const _Float16* w3 = sW + 2048 + 16384;
    v16h bf0 = fragB(w3, n0, HDIM, 0,  lane);
    v16h bf1 = fragB(w3, n0, HDIM, 32, lane);
    v16h bf2 = fragB(w3, n0, HDIM, 64, lane);
    v16h bf3 = fragB(w3, n0, HDIM, 96, lane);
#pragma unroll
    for (int t = 0; t < 10; ++t) {
      v8f acc = {};
      acc = wmma16(fragA(sHb, t * 16, HDIM, 0,  lane), bf0, acc);
      acc = wmma16(fragA(sHb, t * 16, HDIM, 32, lane), bf1, acc);
      acc = wmma16(fragA(sHb, t * 16, HDIM, 64, lane), bf2, acc);
      acc = wmma16(fragA(sHb, t * 16, HDIM, 96, lane), bf3, acc);
      store_bnrelu_h(acc, sHa, t * 16, n0, sST + 4 * HDIM, sST + 5 * HDIM, lane);
    }
  }
  __syncthreads();

  // Export x (f16) for the pair stage.
  copy_half8(x_ws + (size_t)b0 * 10 * HDIM, sHa, 160 * HDIM, tid, 256);

  // Masked aggregation, single pass: sum / max / sumsq -> mean, var.
  {
    int r = tid >> 4, cg = tid & 15;
    float cnt = (float)sJn[r];
#pragma unroll
    for (int cc = 0; cc < 8; ++cc) {
      int c = cg * 8 + cc;
      float sum = 0.0f, sq = 0.0f, mx = -3.0e38f;
      for (int j = 0; j < 10; ++j) {
        if (sMask[r * 10 + j]) {
          float v = (float)sHa[(r * 10 + j) * HDIM + c];
          sum += v; sq += v * v; mx = fmaxf(mx, v);
        }
      }
      float mean = sum / cnt;
      size_t ob = (size_t)(b0 + r) * 512;
      aggx[ob + c]       = sum;
      aggx[ob + 128 + c] = mx;
      aggx[ob + 256 + c] = mean;
      aggx[ob + 384 + c] = sq / cnt - mean * mean;
    }
  }
}

// ---------------------------------------------------------------------------
// MLP2 + agg_y: one WG = 2 batch rows; 2 x 45 pairs padded to M=96 (6 M-tiles).
// Each wave owns one fixed N-tile (nt == wid); final layer aggregates
// in-register and reduces via LDS float-add / uint-max atomics (no Y buffer).
// ---------------------------------------------------------------------------
__global__ __launch_bounds__(256) void k_mlp2(
    const _Float16* __restrict__ wT2, const float* __restrict__ st2,
    const int* __restrict__ jn_ws, const _Float16* __restrict__ x_ws,
    const int* __restrict__ dv, float* __restrict__ aggy)
{
  extern __shared__ char smem[];
  _Float16* sW = (_Float16*)smem;           // 3 * 16384
  _Float16* sA = sW + 49152;                // 96*128
  _Float16* sH = sA + 96 * HDIM;            // 96*128
  _Float16* sX = sH + 96 * HDIM;            // 2*10*128
  float*    sST  = (float*)(sX + 2 * 10 * HDIM); // 768
  float*    sRed = sST + 768;               // 2 rows * [sum|sq|max] * 128
  int*      sDV  = (int*)(sRed + 768);      // 2*90
  int*      sNp  = sDV + 180;               // 2

  int tid = threadIdx.x, lane = tid & 31, wid = tid >> 5;
  int b0 = blockIdx.x * RB2;

  copy_g2l(sW, wT2, 49152, tid, 256);
  copy_g2l(sX, x_ws + (size_t)b0 * 10 * HDIM, RB2 * 10 * HDIM, tid, 256);
  for (int i = tid; i < 768; i += 256) sST[i] = st2[i];
  for (int i = tid; i < 768; i += 256) sRed[i] = 0.0f;
  for (int i = tid; i < 180; i += 256) {
    int rb = i / 90;
    int jn = jn_ws[b0 + rb];
    sDV[i] = dv[(jn - 2) * 90 + (i - rb * 90)];
  }
  if (tid < RB2) {
    int jn = jn_ws[b0 + tid];
    sNp[tid] = (jn * (jn - 1)) >> 1;
  }
  g2l_commit();
  __syncthreads();

  // Build pairs_sum A-matrix (96x128 f16); rows >= npairs are zero.
  for (int e = tid; e < 96 * HDIM; e += 256) {
    int p = e >> 7, c = e & 127;
    int rb = (p >= 48) ? 1 : 0;
    int pl = p - rb * 48;
    _Float16 v = (_Float16)0.0f;
    if (pl < sNp[rb]) {
      int i0 = sDV[rb * 90 + 2 * pl], i1 = sDV[rb * 90 + 2 * pl + 1];
      v = (_Float16)((float)sX[(rb * 10 + i0) * HDIM + c] +
                     (float)sX[(rb * 10 + i1) * HDIM + c]);
    }
    sA[e] = v;
  }
  __syncthreads();

  int n0 = wid * 16;      // this wave's fixed N-tile

  // Layer 0: sA -> sH
  {
    v16h bf0 = fragB(sW, n0, HDIM, 0,  lane);
    v16h bf1 = fragB(sW, n0, HDIM, 32, lane);
    v16h bf2 = fragB(sW, n0, HDIM, 64, lane);
    v16h bf3 = fragB(sW, n0, HDIM, 96, lane);
#pragma unroll
    for (int mt = 0; mt < 6; ++mt) {
      v8f acc = {};
      acc = wmma16(fragA(sA, mt * 16, HDIM, 0,  lane), bf0, acc);
      acc = wmma16(fragA(sA, mt * 16, HDIM, 32, lane), bf1, acc);
      acc = wmma16(fragA(sA, mt * 16, HDIM, 64, lane), bf2, acc);
      acc = wmma16(fragA(sA, mt * 16, HDIM, 96, lane), bf3, acc);
      store_bnrelu_h(acc, sH, mt * 16, n0, sST, sST + HDIM, lane);
    }
  }
  __syncthreads();

  // Layer 1: sH -> sA
  {
    const _Float16* w = sW + 16384;
    v16h bf0 = fragB(w, n0, HDIM, 0,  lane);
    v16h bf1 = fragB(w, n0, HDIM, 32, lane);
    v16h bf2 = fragB(w, n0, HDIM, 64, lane);
    v16h bf3 = fragB(w, n0, HDIM, 96, lane);
#pragma unroll
    for (int mt = 0; mt < 6; ++mt) {
      v8f acc = {};
      acc = wmma16(fragA(sH, mt * 16, HDIM, 0,  lane), bf0, acc);
      acc = wmma16(fragA(sH, mt * 16, HDIM, 32, lane), bf1, acc);
      acc = wmma16(fragA(sH, mt * 16, HDIM, 64, lane), bf2, acc);
      acc = wmma16(fragA(sH, mt * 16, HDIM, 96, lane), bf3, acc);
      store_bnrelu_h(acc, sA, mt * 16, n0, sST + 2 * HDIM, sST + 3 * HDIM, lane);
    }
  }
  __syncthreads();

  // Layer 2: sA -> in-register masked aggregation (sum, sumsq, max>=0).
  {
    const _Float16* w = sW + 32768;
    v16h bf0 = fragB(w, n0, HDIM, 0,  lane);
    v16h bf1 = fragB(w, n0, HDIM, 32, lane);
    v16h bf2 = fragB(w, n0, HDIM, 64, lane);
    v16h bf3 = fragB(w, n0, HDIM, 96, lane);
    int n  = n0 + (lane & 15);
    int mb = (lane >> 4) << 3;
    float sc = sST[4 * HDIM + n], bi = sST[5 * HDIM + n];
#pragma unroll
    for (int mt = 0; mt < 6; ++mt) {
      v8f acc = {};
      acc = wmma16(fragA(sA, mt * 16, HDIM, 0,  lane), bf0, acc);
      acc = wmma16(fragA(sA, mt * 16, HDIM, 32, lane), bf1, acc);
      acc = wmma16(fragA(sA, mt * 16, HDIM, 64, lane), bf2, acc);
      acc = wmma16(fragA(sA, mt * 16, HDIM, 96, lane), bf3, acc);
      int rb = (mt >= 3) ? 1 : 0;
      int pb = (mt - 3 * rb) * 16 + mb;
      int np = sNp[rb];
      float sum = 0.0f, sq = 0.0f, mx = 0.0f;   // post-ReLU values are >= 0
#pragma unroll
      for (int i = 0; i < 8; ++i) {
        float v = fmaxf(acc[i] * sc + bi, 0.0f);
        v = ((pb + i) < np) ? v : 0.0f;
        sum += v; sq += v * v; mx = fmaxf(mx, v);
      }
      atomicAdd(&sRed[rb * 384 + n], sum);
      atomicAdd(&sRed[rb * 384 + 128 + n], sq);
      atomicMax((unsigned int*)&sRed[rb * 384 + 256 + n], __float_as_uint(mx));
    }
  }
  __syncthreads();

  // Final stats: 2 rows x 128 cols over 256 threads.
  {
    int r = tid >> 7, c = tid & 127;
    float cnt = (float)sNp[r];
    float sum = sRed[r * 384 + c];
    float sq  = sRed[r * 384 + 128 + c];
    float mx  = sRed[r * 384 + 256 + c];
    float mean = sum / cnt;
    size_t ob = (size_t)(b0 + r) * 512;
    aggy[ob + c]       = sum;
    aggy[ob + 128 + c] = mx;
    aggy[ob + 256 + c] = mean;
    aggy[ob + 384 + c] = sq / cnt - mean * mean;
  }
}

// ---------------------------------------------------------------------------
extern "C" void kernel_launch(void* const* d_in, const int* in_sizes, int n_in,
                              void* d_out, int out_size, void* d_ws, size_t ws_size,
                              hipStream_t stream) {
  (void)in_sizes; (void)n_in; (void)out_size; (void)ws_size;

  const float* jets      = (const float*)d_in[0];
  const float* mean_jets = (const float*)d_in[2];
  const float* std_jets  = (const float*)d_in[3];
  const float* w1_first  = (const float*)d_in[6];
  const float* w1_rest   = (const float*)d_in[7];
  const float* bn1       = (const float*)d_in[8];
  const float* w2_first  = (const float*)d_in[9];
  const float* w2_rest   = (const float*)d_in[10];
  const float* bn2       = (const float*)d_in[11];
  const int*   dict_vals = (const int*)d_in[12];
  float* out = (float*)d_out;

  // Workspace layout
  char* ws = (char*)d_ws;
  _Float16* wT   = (_Float16*)ws;              // 83968 halfs = 167936 B
  float*    st   = (float*)(ws + 167936);      // 1536 floats -> 174080
  int*      jnws = (int*)(ws + 174080);        // 16384 ints  -> 239616
  _Float16* x_ws = (_Float16*)(ws + 239616);   // B*10*128 halfs (~40 MB)

  k_prep<<<334, 256, 0, stream>>>(w1_first, w1_rest, bn1, w2_first, w2_rest, bn2,
                                  wT, st);

  constexpr int SMEM1 = (160 * 16 + 34816 + 2 * 160 * HDIM) * 2   // halfs
                      + 768 * 4 + 160 * 4 + 16 * 4;               // ~160.4 KB
  k_mlp1<<<BATCH / TB, 256, SMEM1, stream>>>(jets, mean_jets, std_jets, wT, st,
                                             jnws, x_ws, out);

  constexpr int SMEM2 = (49152 + 2 * 96 * HDIM + 2 * 10 * HDIM) * 2  // halfs
                      + 768 * 4 + 768 * 4 + 180 * 4 + 2 * 4;         // ~159.4 KB
  k_mlp2<<<BATCH / RB2, 256, SMEM2, stream>>>(wT + 34816, st + 768, jnws, x_ws,
                                              dict_vals, out + (size_t)BATCH * 512);
}
